// TSM_22660247453906
// MI455X (gfx1250) — matrix operand order repaired and checked by reading, not compiled, not run on hardware
//
#include <hip/hip_runtime.h>
#include <hip/hip_bf16.h>
#include <stddef.h>

// ---- problem constants (match reference setup_inputs) ----
#define BATCH 8
#define NQ    2048
#define NS    4096
#define DIM   512
#define EPSN  1e-12f
#define ROWB  1040            // LDS bytes per B-tile row: 1024 data + 16 pad (bank swizzle)
#define TILEB (16 * ROWB)     // one B tile in LDS

typedef __attribute__((ext_vector_type(16))) __bf16 v16bf;
typedef __attribute__((ext_vector_type(8)))  __bf16 bf16x8;
typedef __attribute__((ext_vector_type(4)))  __bf16 bf16x4;
typedef __attribute__((ext_vector_type(8)))  float  v8f;

// ------------------------------------------------------------------
// Pass 0: zero the 8 output accumulators (atomicAdd targets)
// ------------------------------------------------------------------
__global__ void TSM_zero_out(float* __restrict__ out, int n) {
    int i = threadIdx.x;
    if (i < n) out[i] = 0.0f;
}

// ------------------------------------------------------------------
// Pass 1: L2-normalize rows of [nrows, 512] f32 -> bf16
// one wave per row; each lane handles 16 floats (4x float4)
// ------------------------------------------------------------------
__global__ void TSM_normalize_bf16(const float* __restrict__ src,
                                   __bf16* __restrict__ dst,
                                   int nrows) {
    const int wave = threadIdx.x >> 5;
    const int lane = threadIdx.x & 31;
    const int row  = blockIdx.x * 8 + wave;
    if (row >= nrows) return;

    const float4* s = reinterpret_cast<const float4*>(src + (size_t)row * DIM);
    float4 v[4];
    float ss = 0.0f;
#pragma unroll
    for (int j = 0; j < 4; ++j) {
        v[j] = s[lane + j * 32];
        ss += v[j].x * v[j].x + v[j].y * v[j].y + v[j].z * v[j].z + v[j].w * v[j].w;
    }
#pragma unroll
    for (int off = 16; off >= 1; off >>= 1)
        ss += __shfl_xor(ss, off, 32);

    const float scale = 1.0f / fmaxf(sqrtf(ss), EPSN);

    __bf16* d = dst + (size_t)row * DIM;
#pragma unroll
    for (int j = 0; j < 4; ++j) {
        bf16x4 b;
        b.x = (__bf16)(v[j].x * scale);
        b.y = (__bf16)(v[j].y * scale);
        b.z = (__bf16)(v[j].z * scale);
        b.w = (__bf16)(v[j].w * scale);
        *reinterpret_cast<bf16x4*>(d + (size_t)(lane + j * 32) * 4) = b;
    }
}

// ------------------------------------------------------------------
// Pass 2: block = 8 waves, each wave owns a 16-query M-tile; the
// block sweeps all 256 support N-tiles with the B tile staged in
// LDS (double-buffered). Global loads for tile nt+1 are issued
// before the 16-WMMA compute of tile nt and only land in ds_store
// afterwards, so L2 latency is hidden structurally. Inner loop is
// ds_load_b128 -> v_wmma_f32_16x16x32_bf16.
// ------------------------------------------------------------------
__global__ __launch_bounds__(256)
void TSM_cosmin_wmma(const __bf16* __restrict__ qn,
                     const __bf16* __restrict__ sn,
                     float* __restrict__ out) {
    __shared__ __attribute__((aligned(64))) char smem[2 * TILEB];   // 32.5 KB

    const int tid  = threadIdx.x;
    const int wave = tid >> 5;
    const int lane = tid & 31;
    const int half = lane >> 4;        // 0: lanes 0-15, 1: lanes 16-31
    const int l16  = lane & 15;

    const int batch = blockIdx.x >> 4;             // 16 blocks per batch
    const int mtile = (blockIdx.x & 15) * 8 + wave;

    const __bf16* sbase = sn + (size_t)batch * NS * DIM;

    // --- load + cache all 16 A fragments (M row = l16) ---
    // 16-bit A 16x32 layout: half 0 -> K 0..15, half 1 -> K 16..31 (contiguous, matches B)
    const __bf16* qrow = qn + ((size_t)batch * NQ + (size_t)mtile * 16 + l16) * DIM;
    v16bf aw[16];
#pragma unroll
    for (int ks = 0; ks < 16; ++ks) {
        const int kb = ks * 32 + half * 16;
        bf16x8 lo = *reinterpret_cast<const bf16x8*>(qrow + kb);
        bf16x8 hi = *reinterpret_cast<const bf16x8*>(qrow + kb + 8);
        aw[ks] = __builtin_shufflevector(lo, hi,
                    0, 1, 2, 3, 4, 5, 6, 7, 8, 9, 10, 11, 12, 13, 14, 15);
    }

    float vmax[8];
#pragma unroll
    for (int r = 0; r < 8; ++r) vmax[r] = -2.0f;   // sims are in [-1,1]

    // --- staging coordinates: thread t copies 64B of row (t>>4), chunk (t&15)
    const int srow_st = tid >> 4;                  // 0..15
    const int schunk  = tid & 15;                  // 0..15 (64B chunks)
    const __bf16* gsrc = sbase + (size_t)srow_st * DIM + schunk * 32;  // elements
    char* lsta = smem + srow_st * ROWB + schunk * 64;

    // prologue: stage tile 0 into buffer 0
    {
        const float4* g = reinterpret_cast<const float4*>(gsrc);
        float4 r0 = g[0], r1 = g[1], r2 = g[2], r3 = g[3];
        float4* l = reinterpret_cast<float4*>(lsta);
        l[0] = r0; l[1] = r1; l[2] = r2; l[3] = r3;
    }
    __syncthreads();

    const char* lfrag = smem + l16 * ROWB + half * 32;  // B frag base (this lane)

    for (int nt = 0; nt < 256; ++nt) {
        const int cur = nt & 1;

        // 1) issue global loads for tile nt+1 (consumed only at step 3)
        float4 r0, r1, r2, r3;
        const bool more = (nt + 1) < 256;
        if (more) {
            const float4* g = reinterpret_cast<const float4*>(
                gsrc + (size_t)(nt + 1) * 16 * DIM);
            r0 = g[0]; r1 = g[1]; r2 = g[2]; r3 = g[3];
            __builtin_prefetch(gsrc + (size_t)(nt + 2) * 16 * DIM, 0, 3);
        }

        // 2) compute tile nt from LDS: 16 K-steps of bf16 WMMA
        const char* lb = lfrag + cur * TILEB;
        v8f c = {};
#pragma unroll
        for (int ks = 0; ks < 16; ++ks) {
            // B 32x16 layout: half 0 -> K 0..15, half 1 -> K 16..31 (contiguous 32B)
            bf16x8 blo = *reinterpret_cast<const bf16x8*>(lb + ks * 64);
            bf16x8 bhi = *reinterpret_cast<const bf16x8*>(lb + ks * 64 + 16);
            v16bf b = __builtin_shufflevector(blo, bhi,
                        0, 1, 2, 3, 4, 5, 6, 7, 8, 9, 10, 11, 12, 13, 14, 15);
            c = __builtin_amdgcn_wmma_f32_16x16x32_bf16(
                    false, aw[ks], false, b, (short)0, c, false, false);
        }
#pragma unroll
        for (int r = 0; r < 8; ++r) vmax[r] = fmaxf(vmax[r], c[r]);

        // 3) store staged tile nt+1 into the other buffer
        if (more) {
            float4* l = reinterpret_cast<float4*>(lsta + (cur ^ 1) * TILEB);
            l[0] = r0; l[1] = r1; l[2] = r2; l[3] = r3;
        }
        __syncthreads();
    }

    // --- max over the 16 lanes of each half (N direction) ---
#pragma unroll
    for (int off = 8; off >= 1; off >>= 1) {
#pragma unroll
        for (int r = 0; r < 8; ++r)
            vmax[r] = fmaxf(vmax[r], __shfl_xor(vmax[r], off, 32));
    }

    // rows: half 0 holds M=0..7, half 1 holds M=8..15
    float part = 0.0f;
#pragma unroll
    for (int r = 0; r < 8; ++r) part += 1.0f - vmax[r];
    part += __shfl_xor(part, 16, 32);              // combine the two halves

    if (lane == 0)
        atomicAdd(out + batch, part * (1.0f / (float)NQ));
}

// ------------------------------------------------------------------
extern "C" void kernel_launch(void* const* d_in, const int* in_sizes, int n_in,
                              void* d_out, int out_size, void* d_ws, size_t ws_size,
                              hipStream_t stream) {
    const float* query   = (const float*)d_in[0];   // [8,2048,512] f32
    const float* support = (const float*)d_in[1];   // [8,4096,512] f32
    float* out = (float*)d_out;                     // [8] f32

    __bf16* qn = (__bf16*)d_ws;                                   // 16 MB
    __bf16* sn = (__bf16*)((char*)d_ws + (size_t)BATCH * NQ * DIM * sizeof(__bf16)); // 32 MB

    TSM_zero_out<<<1, 64, 0, stream>>>(out, out_size);

    TSM_normalize_bf16<<<(BATCH * NQ) / 8, 256, 0, stream>>>(query, qn, BATCH * NQ);
    TSM_normalize_bf16<<<(BATCH * NS) / 8, 256, 0, stream>>>(support, sn, BATCH * NS);

    // 8 batches * 16 blocks (8 M-tiles each) = 128 blocks of 256 threads
    TSM_cosmin_wmma<<<BATCH * 16, 256, 0, stream>>>(qn, sn, out);
}
